// Head_21672404975739
// MI455X (gfx1250) — matrix operand order repaired and checked by reading, not compiled, not run on hardware
//
#include <hip/hip_runtime.h>
#include <math.h>

// ---------------------------------------------------------------------------
// CDNA5 (gfx1250) flash-attention, single head.
//   B=8, T=4096, C=384, H=64.  fp32 in/out, bf16 WMMA compute (f32 accum).
//   v2: transposed-K LDS layout + double-buffered tiles.
// ---------------------------------------------------------------------------

typedef __attribute__((ext_vector_type(16))) __bf16 v16bf;
typedef __attribute__((ext_vector_type(8)))  float  v8f;

#define EMBED  384
#define SEQ    4096
#define HEAD   64
#define NBATCH 8
#define KTILE  64
#define ROWS_PER_BLOCK 128   // 8 waves * 16 rows

static __device__ __forceinline__ v8f wmma_bf16(v16bf a, v16bf b, v8f c) {
  // v_wmma_f32_16x16x32_bf16  (neg_a, A, neg_b, B, c_mod, C, reuse_a, reuse_b)
  return __builtin_amdgcn_wmma_f32_16x16x32_bf16(false, a, false, b, (short)0, c,
                                                 false, false);
}

// ---------------------------------------------------------------------------
// Projection: out[row][h] = scale * sum_c x[row][c] * W[c][h]   (bf16 output)
// One block = 128 rows, 8 waves, each wave owns 16 rows (WMMA M).
// ---------------------------------------------------------------------------
__global__ __launch_bounds__(256)
void proj_bf16_wmma(const float* __restrict__ x, const float* __restrict__ W,
                    __bf16* __restrict__ outp, float scale)
{
  __shared__ __bf16 xt[128][40];            // 32-wide K slab, +8 pad
  const int tid  = threadIdx.x;
  const int wave = tid >> 5;
  const int lane = tid & 31;
  const int lr   = lane & 15;
  const int g    = lane >> 4;
  const size_t row0 = (size_t)blockIdx.x * 128;

  v8f z = {0.f,0.f,0.f,0.f,0.f,0.f,0.f,0.f};
  v8f acc[4] = {z, z, z, z};                // 4 head sub-tiles of 16

  for (int c0 = 0; c0 < EMBED; c0 += 32) {
    __syncthreads();
    // stage x[row0..row0+127][c0..c0+31] as bf16 into LDS
#pragma unroll
    for (int e = tid; e < 128 * 8; e += 256) {
      int row = e >> 3, seg = e & 7;
      float4 xv = *(const float4*)&x[(row0 + row) * EMBED + c0 + seg * 4];
      xt[row][seg * 4 + 0] = (__bf16)xv.x;
      xt[row][seg * 4 + 1] = (__bf16)xv.y;
      xt[row][seg * 4 + 2] = (__bf16)xv.z;
      xt[row][seg * 4 + 3] = (__bf16)xv.w;
    }
    __syncthreads();

    // A fragment: 16x32 bf16, lane = row, ISA K packing
    v16bf a;
#pragma unroll
    for (int i = 0; i < 16; ++i) {
      int k = (i < 8) ? (g * 8 + i) : (8 + g * 8 + i);
      a[i] = xt[wave * 16 + lr][k];
    }
    // B fragments: W slice 32 x 16 per head sub-tile (lane = N column, elements = K, ISA K packing)
#pragma unroll
    for (int h = 0; h < 4; ++h) {
      v16bf bw;
      int kc = c0;
#pragma unroll
      for (int i = 0; i < 16; ++i)
        bw[i] = (__bf16)W[(kc + ((i < 8) ? (g * 8 + i) : (8 + g * 8 + i))) * HEAD + h * 16 + lr];
      acc[h] = wmma_bf16(a, bw, acc[h]);
    }
  }

  __bf16* op = outp + (row0 + (size_t)wave * 16) * HEAD;
#pragma unroll
  for (int h = 0; h < 4; ++h)
#pragma unroll
    for (int r = 0; r < 8; ++r)
      op[(r + 8 * g) * HEAD + h * 16 + lr] = (__bf16)((float)acc[h][r] * scale);
}

// ---------------------------------------------------------------------------
// Flash attention. Block = (row block of 128, batch). 8 waves x 16 rows.
// Double-buffered 64-key K/V tiles in LDS (K stored TRANSPOSED as [head][key]; Q*K^T
// B-fragment elements step along head rows); online softmax; bf16 WMMA GEMMs.
// ---------------------------------------------------------------------------
__global__ __launch_bounds__(256)
void attn_flash_wmma(const __bf16* __restrict__ qg, const __bf16* __restrict__ kg,
                     const __bf16* __restrict__ vg, float* __restrict__ out)
{
  __shared__ __bf16 ktT[2][HEAD][KTILE + 8];    // K^T: [head][key], 18.4 KB
  __shared__ __bf16 vt [2][KTILE][HEAD + 8];    // V row-major,      18.4 KB
  __shared__ __bf16 pbuf[8][16][KTILE + 8];     // per-wave P stage, 18.4 KB

  const int tid  = threadIdx.x;
  const int wave = tid >> 5;
  const int lane = tid & 31;
  const int lr   = lane & 15;   // A: row-in-16 / B: column-in-16 / C: column
  const int g    = lane >> 4;

  const int b  = blockIdx.y;
  const int m0 = blockIdx.x * ROWS_PER_BLOCK;
  const int mw = m0 + wave * 16;                // this wave's first query row

  const __bf16* kbase = kg + (size_t)b * SEQ * HEAD;
  const __bf16* vbase = vg + (size_t)b * SEQ * HEAD;

  // ---- load Q fragments (rows mw..mw+15, head 0..63), scale pre-folded ----
  const __bf16* qrow = qg + ((size_t)b * SEQ + mw) * HEAD;
  v16bf aq[2];
#pragma unroll
  for (int f = 0; f < 2; ++f)
#pragma unroll
    for (int i = 0; i < 16; ++i) {
      int k = (i < 8) ? (g * 8 + i) : (8 + g * 8 + i);
      aq[f][i] = qrow[lr * HEAD + f * 32 + k];
    }

  v8f z = {0.f,0.f,0.f,0.f,0.f,0.f,0.f,0.f};
  v8f acc[4] = {z, z, z, z};                    // O accumulator 16 x 64
  float m_i[8], l_i[8];
#pragma unroll
  for (int r = 0; r < 8; ++r) { m_i[r] = -3.0e38f; l_i[r] = 0.0f; }

  const int jend = m0 + ROWS_PER_BLOCK;         // causal: keys < jend
  const int nt   = jend / KTILE;

  // ---- prologue: stage tile 0 into buffer 0 (K transposed, V row-major) ----
#pragma unroll
  for (int e = tid; e < KTILE * 8; e += 256) {
    int row = e >> 3, seg = e & 7;
    uint4 kv = *(const uint4*)&kbase[row * HEAD + seg * 8];
    *(uint4*)&vt[0][row][seg * 8] = *(const uint4*)&vbase[row * HEAD + seg * 8];
    __bf16 ktmp[8];
    *(uint4*)ktmp = kv;
#pragma unroll
    for (int c = 0; c < 8; ++c)
      ktT[0][seg * 8 + c][row] = ktmp[c];
  }
  __syncthreads();

  for (int t = 0; t < nt; ++t) {
    const int j0  = t * KTILE;
    const int buf = t & 1;

    // ---- stage next tile into the other buffer (overlaps with compute) ----
    if (t + 1 < nt) {
      const __bf16* kgp = kbase + (size_t)(j0 + KTILE) * HEAD;
      const __bf16* vgp = vbase + (size_t)(j0 + KTILE) * HEAD;
#pragma unroll
      for (int e = tid; e < KTILE * 8; e += 256) {
        int row = e >> 3, seg = e & 7;
        uint4 kv = *(const uint4*)&kgp[row * HEAD + seg * 8];
        *(uint4*)&vt[buf ^ 1][row][seg * 8] = *(const uint4*)&vgp[row * HEAD + seg * 8];
        __bf16 ktmp[8];
        *(uint4*)ktmp = kv;
#pragma unroll
        for (int c = 0; c < 8; ++c)
          ktT[buf ^ 1][seg * 8 + c][row] = ktmp[c];
      }
      if (j0 + 2 * KTILE < jend) {               // global_prefetch_b8 tile t+2
        __builtin_prefetch(kbase + (size_t)(j0 + 2 * KTILE) * HEAD, 0, 1);
        __builtin_prefetch(vbase + (size_t)(j0 + 2 * KTILE) * HEAD, 0, 1);
      }
    }

    if (j0 <= mw) {   // tiles past this wave's rows are fully masked: skip
      // -------- S = Q K^T  (16 rows x 64 keys) --------
      v8f s[4];
#pragma unroll
      for (int n = 0; n < 4; ++n) {
        v8f sc = z;
#pragma unroll
        for (int f = 0; f < 2; ++f) {
          v16bf bk;                 // B: (head 32) x (key 16); lane = key, elements = head (ISA K packing)
          int c = f * 32;
#pragma unroll
          for (int i = 0; i < 16; ++i)
            bk[i] = ktT[buf][c + ((i < 8) ? (g * 8 + i) : (8 + g * 8 + i))][n * 16 + lr];
          sc = wmma_bf16(aq[f], bk, sc);
        }
        s[n] = sc;
      }

      // -------- causal mask + row max --------
      float tmax[8];
#pragma unroll
      for (int r = 0; r < 8; ++r) tmax[r] = -3.0e38f;
#pragma unroll
      for (int n = 0; n < 4; ++n) {
        int j = j0 + n * 16 + lr;                // key index (C-layout column)
#pragma unroll
        for (int r = 0; r < 8; ++r) {
          int m = mw + r + 8 * g;                // query row
          float sv = s[n][r];
          if (j > m) sv = -3.0e38f;
          s[n][r] = sv;
          tmax[r] = fmaxf(tmax[r], sv);
        }
      }
#pragma unroll
      for (int off = 1; off < 16; off <<= 1)     // butterfly within 16-lane group
#pragma unroll
        for (int r = 0; r < 8; ++r)
          tmax[r] = fmaxf(tmax[r], __shfl_xor(tmax[r], off, 32));

      // -------- online softmax update --------
      float alpha[8], rsum[8];
#pragma unroll
      for (int r = 0; r < 8; ++r) {
        float mn = fmaxf(m_i[r], tmax[r]);
        alpha[r] = __expf(m_i[r] - mn);
        m_i[r] = mn;
        rsum[r] = 0.0f;
      }
#pragma unroll
      for (int n = 0; n < 4; ++n)
#pragma unroll
        for (int r = 0; r < 8; ++r) {
          float p = __expf(s[n][r] - m_i[r]);
          s[n][r] = p;
          rsum[r] += p;
        }
#pragma unroll
      for (int off = 1; off < 16; off <<= 1)
#pragma unroll
        for (int r = 0; r < 8; ++r)
          rsum[r] += __shfl_xor(rsum[r], off, 32);
#pragma unroll
      for (int r = 0; r < 8; ++r)
        l_i[r] = l_i[r] * alpha[r] + rsum[r];
#pragma unroll
      for (int h = 0; h < 4; ++h)
#pragma unroll
        for (int r = 0; r < 8; ++r)
          acc[h][r] *= alpha[r];

      // -------- reshape P (C-layout -> A-layout) via per-wave LDS --------
#pragma unroll
      for (int n = 0; n < 4; ++n)
#pragma unroll
        for (int r = 0; r < 8; ++r)
          pbuf[wave][r + 8 * g][n * 16 + lr] = (__bf16)s[n][r];
      // compiler inserts the required s_wait_dscnt for the aliasing reload

      v16bf ap[2];
#pragma unroll
      for (int f = 0; f < 2; ++f)
#pragma unroll
        for (int i = 0; i < 16; ++i) {
          int k = (i < 8) ? (g * 8 + i) : (8 + g * 8 + i);
          ap[f][i] = pbuf[wave][lr][f * 32 + k];
        }

      // -------- O += P V  (16 x 64 keys  x  64 head) --------
#pragma unroll
      for (int h = 0; h < 4; ++h) {
        v8f oc = acc[h];
#pragma unroll
        for (int f = 0; f < 2; ++f) {
          v16bf bv;                              // B: (key 32) x (head 16)
          int kr = f * 32;
#pragma unroll
          for (int i = 0; i < 16; ++i)
            bv[i] = vt[buf][kr + ((i < 8) ? (g * 8 + i) : (8 + g * 8 + i))][h * 16 + lr];
          oc = wmma_bf16(ap[f], bv, oc);
        }
        acc[h] = oc;
      }
    }

    __syncthreads();   // all waves done with buf before it is restaged
  }

  // -------- normalize and store fp32 output --------
  float* op = out + ((size_t)b * SEQ + mw) * HEAD;
#pragma unroll
  for (int h = 0; h < 4; ++h)
#pragma unroll
    for (int r = 0; r < 8; ++r)
      op[(r + 8 * g) * HEAD + h * 16 + lr] = (float)acc[h][r] / l_i[r];
}

// ---------------------------------------------------------------------------
extern "C" void kernel_launch(void* const* d_in, const int* in_sizes, int n_in,
                              void* d_out, int out_size, void* d_ws, size_t ws_size,
                              hipStream_t stream)
{
  const float* x  = (const float*)d_in[0];
  const float* Wk = (const float*)d_in[1];
  const float* Wq = (const float*)d_in[2];
  const float* Wv = (const float*)d_in[3];
  float* out = (float*)d_out;

  const size_t nqkv = (size_t)NBATCH * SEQ * HEAD;      // 2M elems, bf16
  __bf16* qws = (__bf16*)d_ws;
  __bf16* kws = qws + nqkv;
  __bf16* vws = kws + nqkv;

  const int proj_blocks = (NBATCH * SEQ) / 128;          // 256
  const float qscale = 1.0f / sqrtf((float)EMBED);       // reference: /sqrt(C)

  proj_bf16_wmma<<<proj_blocks, 256, 0, stream>>>(x, Wq, qws, qscale);
  proj_bf16_wmma<<<proj_blocks, 256, 0, stream>>>(x, Wk, kws, 1.0f);
  proj_bf16_wmma<<<proj_blocks, 256, 0, stream>>>(x, Wv, vws, 1.0f);

  dim3 grid(SEQ / ROWS_PER_BLOCK, NBATCH);               // (32, 8)
  attn_flash_wmma<<<grid, 256, 0, stream>>>(qws, kws, vws, out);
}
